// SpatialGroupingAttention_59365037965790
// MI455X (gfx1250) — matrix operand order repaired and checked by reading, not compiled, not run on hardware
//
#include <hip/hip_runtime.h>
#include <hip/hip_bf16.h>
#include <math.h>

// ---------------------------------------------------------------------------
// CDNA5 (gfx1250) wave32 WMMA helpers: v_wmma_f32_16x16x32_bf16
// ---------------------------------------------------------------------------
typedef __attribute__((ext_vector_type(16))) __bf16 v16bf;
typedef __attribute__((ext_vector_type(8)))  float  v8f;

__device__ __forceinline__ v8f wmma_bf16(v16bf a, v16bf b, v8f c) {
  return __builtin_amdgcn_wmma_f32_16x16x32_bf16(false, a, false, b, (short)0, c,
                                                 false, false);
}

__device__ __forceinline__ v8f zero_acc() {
  v8f a;
#pragma unroll
  for (int i = 0; i < 8; ++i) a[i] = 0.0f;
  return a;
}

__device__ __forceinline__ v16bf zero_abf() {
  v16bf r;
#pragma unroll
  for (int i = 0; i < 16; ++i) r[i] = (__bf16)0.0f;
  return r;
}

// A fragment (16x32 bf16): lane&15 selects the row; K chunks per ISA layout:
// lanes 0-15 -> K {k0..k0+7, k0+16..k0+23}; lanes 16-31 -> {k0+8..15, k0+24..31}
__device__ __forceinline__ v16bf load_a_frag(const __bf16* rowbase, int k0, int lane) {
  union { uint4 u[2]; v16bf v; } t;
  const int hi8 = (lane & 16) ? 8 : 0;
  t.u[0] = *reinterpret_cast<const uint4*>(rowbase + k0 + hi8);
  t.u[1] = *reinterpret_cast<const uint4*>(rowbase + k0 + 16 + hi8);
  return t.v;
}

// B fragment from transposed operand bt[n][k] (row-major N x K, leading dim ldb):
// lane&15 -> N; lanes 0-15 get K {k0..k0+7, k0+16..k0+23}, lanes 16-31 get K {k0+8..15, k0+24..31}.
__device__ __forceinline__ v16bf load_b_frag(const __bf16* bt, int n0, int k0,
                                             int ldb, int lane) {
  const __bf16* p = bt + (size_t)(n0 + (lane & 15)) * ldb + k0 + ((lane & 16) ? 8 : 0);
  union { uint4 u[2]; v16bf v; } t;
  t.u[0] = *reinterpret_cast<const uint4*>(p);
  t.u[1] = *reinterpret_cast<const uint4*>(p + 16);
  return t.v;
}

// A fragment built from an fp32 row, converted to bf16 on the fly.
__device__ __forceinline__ v16bf load_a_frag_f32(const float* rowbase, int k0, int lane) {
  const int hi8 = (lane & 16) ? 8 : 0;
  const float* p1 = rowbase + k0 + hi8;
  const float* p2 = rowbase + k0 + 16 + hi8;
  v16bf r;
#pragma unroll
  for (int i = 0; i < 8; ++i) { r[i] = (__bf16)p1[i]; r[8 + i] = (__bf16)p2[i]; }
  return r;
}

// ---------------------------------------------------------------------------
// Problem constants
// ---------------------------------------------------------------------------
#define B_    2
#define NIN   4096     // 64*64
#define D_    384
#define H_    12
#define NOUT  484      // 22*22
#define MQ    968      // B_*NOUT
#define BH    24       // B_*H_
#define NQPAD 496      // 31*16
#define ATTN_ELEMS 47579136   // BH*NOUT*4096
#define EPS_  1.1920929e-07f

// ---------------------------------------------------------------------------
// Prep / convert kernels
// ---------------------------------------------------------------------------
__global__ void k_cvt_bf16(const float* __restrict__ s, __bf16* __restrict__ d, int n) {
  int i = blockIdx.x * blockDim.x + threadIdx.x;
  if (i < n) d[i] = (__bf16)s[i];
}

// (K,N) fp32 -> (N,K) bf16
__global__ void k_transpose_cvt(const float* __restrict__ s, __bf16* __restrict__ d,
                                int K, int N) {
  int i = blockIdx.x * blockDim.x + threadIdx.x;
  if (i >= K * N) return;
  int k = i / N, n = i % N;
  d[(size_t)n * K + k] = (__bf16)s[i];
}

// conv_w (O=384, I=384, 7,7) -> wq[p][o][i] bf16 (per-position transposed B)
__global__ void k_convw(const float* __restrict__ s, __bf16* __restrict__ d) {
  int i = blockIdx.x * blockDim.x + threadIdx.x;
  if (i >= 384 * 384 * 49) return;
  int p = i % 49;
  int c = (i / 49) % 384;
  int o = i / (49 * 384);
  d[((size_t)p * 384 + o) * 384 + c] = (__bf16)s[i];
}

__global__ void k_copy_f32(const float* __restrict__ s, float* __restrict__ d, int n) {
  int i = blockIdx.x * blockDim.x + threadIdx.x;
  if (i < n) d[i] = s[i];
}

// ---------------------------------------------------------------------------
// Conv as implicit GEMM: M=968 output pixels, N=384, K=49*384.
// Each wave: 16x64 output tile -> 4 accumulators; A fragment reused 4x.
// ---------------------------------------------------------------------------
__global__ void k_gemm_conv(const __bf16* __restrict__ xb, const __bf16* __restrict__ wq,
                            const float* __restrict__ bias, float* __restrict__ xo) {
  const int NG = 6, MT = 61;                 // 384/64 N-groups
  int wave = blockIdx.x * (blockDim.x >> 5) + (threadIdx.x >> 5);
  if (wave >= MT * NG) return;
  int tm = wave / NG, tn = wave % NG;
  int lane = threadIdx.x & 31;
  int m0 = tm * 16, n0 = tn * 64;
  int mrow = m0 + (lane & 15);
  bool mval = mrow < MQ;
  int mm = mval ? mrow : 0;
  int b = mm / NOUT, pix = mm % NOUT;
  int oy = pix / 22, ox = pix % 22;
  v8f acc[4];
#pragma unroll
  for (int j = 0; j < 4; ++j) acc[j] = zero_acc();
  v16bf zf = zero_abf();
  const __bf16* bt = wq;
  for (int kh = 0; kh < 7; ++kh) {
    int iy = oy * 3 - 3 + kh;
    bool yval = mval && iy >= 0 && iy < 64;
    int iyc = min(max(iy, 0), 63);
    for (int kw = 0; kw < 7; ++kw, bt += 384 * 384) {
      int ix = ox * 3 - 3 + kw;
      bool valid = yval && ix >= 0 && ix < 64;
      int ixc = min(max(ix, 0), 63);
      const __bf16* arow = xb + ((size_t)b * NIN + iyc * 64 + ixc) * D_;
#pragma unroll 2
      for (int k = 0; k < D_; k += 32) {
        v16bf a = valid ? load_a_frag(arow, k, lane) : zf;
#pragma unroll
        for (int j = 0; j < 4; ++j)
          acc[j] = wmma_bf16(a, load_b_frag(bt, n0 + j * 16, k, D_, lane), acc[j]);
      }
    }
  }
  int col = lane & 15, rb = (lane & 16) ? 8 : 0;
#pragma unroll
  for (int j = 0; j < 4; ++j) {
    float bv = bias[n0 + j * 16 + col];
#pragma unroll
    for (int r = 0; r < 8; ++r) {
      int m = m0 + rb + r;
      if (m < MQ) xo[(size_t)m * D_ + n0 + j * 16 + col] = acc[j][r] + bv;
    }
  }
}

// ---------------------------------------------------------------------------
// kv = x @ kv_w + kv_b : M=8192, N=768, K=384 (computed once; x is constant)
// 16x64 tile per wave.
// ---------------------------------------------------------------------------
__global__ void k_gemm_kv(const __bf16* __restrict__ xb, const __bf16* __restrict__ bt,
                          const float* __restrict__ bias, float* __restrict__ out) {
  const int NG = 12, MT = 512;               // 768/64 N-groups
  int wave = blockIdx.x * (blockDim.x >> 5) + (threadIdx.x >> 5);
  if (wave >= MT * NG) return;
  int tm = wave / NG, tn = wave % NG;
  int lane = threadIdx.x & 31;
  int m0 = tm * 16, n0 = tn * 64;
  const __bf16* arow = xb + (size_t)(m0 + (lane & 15)) * D_;
  __builtin_prefetch(arow, 0, 1);
  v8f acc[4];
#pragma unroll
  for (int j = 0; j < 4; ++j) acc[j] = zero_acc();
#pragma unroll 2
  for (int k = 0; k < D_; k += 32) {
    v16bf a = load_a_frag(arow, k, lane);
#pragma unroll
    for (int j = 0; j < 4; ++j)
      acc[j] = wmma_bf16(a, load_b_frag(bt, n0 + j * 16, k, D_, lane), acc[j]);
  }
  int col = lane & 15, rb = (lane & 16) ? 8 : 0;
#pragma unroll
  for (int j = 0; j < 4; ++j) {
    float bv = bias[n0 + j * 16 + col];
#pragma unroll
    for (int r = 0; r < 8; ++r)
      out[(size_t)(m0 + rb + r) * 768 + n0 + j * 16 + col] = acc[j][r] + bv;
  }
}

// ---------------------------------------------------------------------------
// Generic row-guarded GEMM (M=968): A (968 x KDIM) bf16, B^T (NDIM x KDIM) bf16
// out fp32 (968 x NDIM) + bias.  16x64 tile per wave.
// ---------------------------------------------------------------------------
template <int NDIM, int KDIM>
__global__ void k_gemm_rows(const __bf16* __restrict__ A, const __bf16* __restrict__ bt,
                            const float* __restrict__ bias, float* __restrict__ out) {
  const int NG = NDIM / 64, MT = 61;
  int wave = blockIdx.x * (blockDim.x >> 5) + (threadIdx.x >> 5);
  if (wave >= MT * NG) return;
  int tm = wave / NG, tn = wave % NG;
  int lane = threadIdx.x & 31;
  int m0 = tm * 16, n0 = tn * 64;
  int mr = m0 + (lane & 15);
  bool av = mr < MQ;
  const __bf16* arow = A + (size_t)(av ? mr : 0) * KDIM;
  __builtin_prefetch(arow, 0, 1);
  v8f acc[4];
#pragma unroll
  for (int j = 0; j < 4; ++j) acc[j] = zero_acc();
  v16bf zf = zero_abf();
  for (int k = 0; k < KDIM; k += 32) {
    v16bf a = av ? load_a_frag(arow, k, lane) : zf;
#pragma unroll
    for (int j = 0; j < 4; ++j)
      acc[j] = wmma_bf16(a, load_b_frag(bt, n0 + j * 16, k, KDIM, lane), acc[j]);
  }
  int col = lane & 15, rb = (lane & 16) ? 8 : 0;
#pragma unroll
  for (int j = 0; j < 4; ++j) {
    float bv = bias[n0 + j * 16 + col];
#pragma unroll
    for (int r = 0; r < 8; ++r) {
      int m = m0 + rb + r;
      if (m < MQ) out[(size_t)m * NDIM + n0 + j * 16 + col] = acc[j][r] + bv;
    }
  }
}

// MLP1 with exact-GELU epilogue, writes bf16 (feeds the second GEMM).
__global__ void k_gemm_mlp1(const __bf16* __restrict__ A, const __bf16* __restrict__ bt,
                            const float* __restrict__ bias, __bf16* __restrict__ h1) {
  const int NG = 24, MT = 61;                // 1536/64
  int wave = blockIdx.x * (blockDim.x >> 5) + (threadIdx.x >> 5);
  if (wave >= MT * NG) return;
  int tm = wave / NG, tn = wave % NG;
  int lane = threadIdx.x & 31;
  int m0 = tm * 16, n0 = tn * 64;
  int mr = m0 + (lane & 15);
  bool av = mr < MQ;
  const __bf16* arow = A + (size_t)(av ? mr : 0) * D_;
  __builtin_prefetch(arow, 0, 1);
  v8f acc[4];
#pragma unroll
  for (int j = 0; j < 4; ++j) acc[j] = zero_acc();
  v16bf zf = zero_abf();
#pragma unroll 2
  for (int k = 0; k < D_; k += 32) {
    v16bf a = av ? load_a_frag(arow, k, lane) : zf;
#pragma unroll
    for (int j = 0; j < 4; ++j)
      acc[j] = wmma_bf16(a, load_b_frag(bt, n0 + j * 16, k, D_, lane), acc[j]);
  }
  int col = lane & 15, rb = (lane & 16) ? 8 : 0;
#pragma unroll
  for (int j = 0; j < 4; ++j) {
    float bv = bias[n0 + j * 16 + col];
#pragma unroll
    for (int r = 0; r < 8; ++r) {
      int m = m0 + rb + r;
      if (m < MQ) {
        float v = acc[j][r] + bv;
        float ge = 0.5f * v * (1.0f + erff(v * 0.70710678f));
        h1[(size_t)m * 1536 + n0 + j * 16 + col] = (__bf16)ge;
      }
    }
  }
}

// ---------------------------------------------------------------------------
// S = q @ k^T per (b,h): K=32 -> one WMMA per 16x16 tile; A reused over 8 N-tiles
// ---------------------------------------------------------------------------
__global__ void k_gemm_qk(const __bf16* __restrict__ qb, const __bf16* __restrict__ kb,
                          float* __restrict__ attn) {
  int wave = blockIdx.x * (blockDim.x >> 5) + (threadIdx.x >> 5);
  if (wave >= BH * 31 * 32) return;
  int bh = wave / (31 * 32);
  int rem = wave % (31 * 32);
  int tm = rem / 32, tg = rem % 32;
  int lane = threadIdx.x & 31;
  const __bf16* qrow = qb + ((size_t)bh * NQPAD + tm * 16 + (lane & 15)) * 32;
  v16bf a = load_a_frag(qrow, 0, lane);
  const __bf16* kbase = kb + (size_t)bh * NIN * 32;
  int col = lane & 15, rb = (lane & 16) ? 8 : 0;
#pragma unroll
  for (int j = 0; j < 8; ++j) {
    int n0 = (tg * 8 + j) * 16;
    v16bf bb = load_b_frag(kbase, n0, 0, 32, lane);
    v8f acc = zero_acc();
    acc = wmma_bf16(a, bb, acc);
#pragma unroll
    for (int r = 0; r < 8; ++r) {
      int m = tm * 16 + rb + r;
      if (m < NOUT) attn[((size_t)bh * NOUT + m) * NIN + n0 + col] = acc[r];
    }
  }
}

// ---------------------------------------------------------------------------
// upd = attn_q @ v per (b,h): M=496(484), N=32, K=4096.
// fp32 probs -> bf16 fragments on the fly; one A fragment feeds both N tiles.
// ---------------------------------------------------------------------------
__global__ void k_gemm_pv(const float* __restrict__ aq, const __bf16* __restrict__ vt,
                          float* __restrict__ upd) {
  int wave = blockIdx.x * (blockDim.x >> 5) + (threadIdx.x >> 5);
  if (wave >= BH * 31) return;
  int bh = wave / 31, tm = wave % 31;
  int lane = threadIdx.x & 31;
  const float* arow = aq + ((size_t)bh * NOUT + tm * 16 + (lane & 15)) * NIN;
  const __bf16* bt = vt + (size_t)bh * 32 * NIN;
  __builtin_prefetch(arow, 0, 1);
  v8f acc0 = zero_acc(), acc1 = zero_acc();
  for (int k = 0; k < NIN; k += 32) {
    if (k + 64 < NIN) __builtin_prefetch(arow + k + 64, 0, 1);
    v16bf a = load_a_frag_f32(arow, k, lane);
    acc0 = wmma_bf16(a, load_b_frag(bt, 0, k, NIN, lane), acc0);
    acc1 = wmma_bf16(a, load_b_frag(bt, 16, k, NIN, lane), acc1);
  }
  int b = bh / H_, h = bh % H_;
  int col = lane & 15, rb = (lane & 16) ? 8 : 0;
#pragma unroll
  for (int r = 0; r < 8; ++r) {
    int m = tm * 16 + rb + r;
    if (m < NOUT) {
      size_t o = ((size_t)(b * NOUT + m)) * D_ + h * 32;
      upd[o + col] = acc0[r];
      upd[o + 16 + col] = acc1[r];
    }
  }
}

// ---------------------------------------------------------------------------
// LayerNorm over D=384 (one row / block, 128 threads, LDS reduction).
// residual!=0: xout += LN(src); always refreshes the bf16 mirror.
// ---------------------------------------------------------------------------
__global__ void k_ln(const float* __restrict__ src, float* __restrict__ xout,
                     __bf16* __restrict__ xoutb, const float* __restrict__ g,
                     const float* __restrict__ bta, int residual) {
  __shared__ float red[128];
  int row = blockIdx.x, t = threadIdx.x;
  const float* p = src + (size_t)row * D_;
  float a0 = p[t], a1 = p[t + 128], a2 = p[t + 256];
  red[t] = a0 + a1 + a2;
  __syncthreads();
  for (int off = 64; off > 0; off >>= 1) {
    if (t < off) red[t] += red[t + off];
    __syncthreads();
  }
  float mu = red[0] * (1.0f / 384.0f);
  __syncthreads();
  float d0 = a0 - mu, d1 = a1 - mu, d2 = a2 - mu;
  red[t] = d0 * d0 + d1 * d1 + d2 * d2;
  __syncthreads();
  for (int off = 64; off > 0; off >>= 1) {
    if (t < off) red[t] += red[t + off];
    __syncthreads();
  }
  float rstd = rsqrtf(red[0] * (1.0f / 384.0f) + 1e-5f);
#pragma unroll
  for (int j = 0; j < 3; ++j) {
    int i = t + j * 128;
    float v = (p[i] - mu) * rstd * g[i] + bta[i];
    size_t oi = (size_t)row * D_ + i;
    float o = residual ? (xout[oi] + v) : v;
    xout[oi] = o;
    xoutb[oi] = (__bf16)o;
  }
}

// ---------------------------------------------------------------------------
// RoSE rotary embedding: K path (64x64 grid, spacing 1, *temp) -> kb + vt
// ---------------------------------------------------------------------------
__global__ void k_rose_kv(const float* __restrict__ kv, const float* __restrict__ freqs,
                          const float* __restrict__ temp, __bf16* __restrict__ kb,
                          __bf16* __restrict__ vt) {
  int blk = blockIdx.x;               // B_*NIN
  int b = blk / NIN, n = blk % NIN;
  int h = threadIdx.x >> 5, d = threadIdx.x & 31;
  const float* base = kv + (size_t)blk * 768;
  float outv = base[h * 32 + d];
  if (d < 16) {
    int j = d >> 1, s = j >> 2, pp = j & 3;
    float coord = (s == 0) ? (float)(n >> 6) : (float)(n & 63);
    float ang = coord * freqs[(h * 2 + s) * 4 + pp];
    float cs = cosf(ang), sn = sinf(ang);
    float x1 = base[h * 32 + j * 2], x2 = base[h * 32 + j * 2 + 1];
    outv = (d & 1) ? (x1 * sn + x2 * cs) : (x1 * cs - x2 * sn);
  }
  outv *= temp[0];
  size_t bh = (size_t)(b * H_ + h);
  kb[(bh * NIN + (size_t)n) * 32 + d] = (__bf16)outv;
  vt[(bh * 32 + (size_t)d) * NIN + n] = (__bf16)base[384 + h * 32 + d];
}

// Q path (22x22 grid, spacing 3) -> qb (padded to 496 rows, pad rows zeroed)
__global__ void k_rose_q(const float* __restrict__ qlin, const float* __restrict__ freqs,
                         __bf16* __restrict__ qb) {
  int blk = blockIdx.x;               // B_*NQPAD
  int b = blk / NQPAD, n = blk % NQPAD;
  int h = threadIdx.x >> 5, d = threadIdx.x & 31;
  size_t bh = (size_t)(b * H_ + h);
  size_t oi = (bh * NQPAD + (size_t)n) * 32 + d;
  if (n >= NOUT) { qb[oi] = (__bf16)0.0f; return; }
  const float* row = qlin + ((size_t)(b * NOUT + n)) * D_ + h * 32;
  float outv = row[d];
  if (d < 16) {
    int j = d >> 1, s = j >> 2, pp = j & 3;
    float coord = (s == 0) ? (float)(n / 22) * 3.0f : (float)(n % 22) * 3.0f;
    float ang = coord * freqs[(h * 2 + s) * 4 + pp];
    float cs = cosf(ang), sn = sinf(ang);
    float x1 = row[j * 2], x2 = row[j * 2 + 1];
    outv = (d & 1) ? (x1 * sn + x2 * cs) : (x1 * cs - x2 * sn);
  }
  qb[oi] = (__bf16)outv;
}

// ---------------------------------------------------------------------------
// Softmax over the key axis (row length 4096), in-place on attn_k
// ---------------------------------------------------------------------------
__global__ void k_softmax(float* __restrict__ attn) {
  __shared__ float red[256];
  int row = blockIdx.x, t = threadIdx.x;
  float* p = attn + (size_t)row * NIN;
  float v[16];
  float mx = -3.4e38f;
#pragma unroll
  for (int j = 0; j < 16; ++j) { v[j] = p[t + j * 256]; mx = fmaxf(mx, v[j]); }
  red[t] = mx;
  __syncthreads();
  for (int off = 128; off > 0; off >>= 1) {
    if (t < off) red[t] = fmaxf(red[t], red[t + off]);
    __syncthreads();
  }
  mx = red[0];
  __syncthreads();
  float s = 0.0f;
#pragma unroll
  for (int j = 0; j < 16; ++j) { v[j] = __expf(v[j] - mx); s += v[j]; }
  red[t] = s;
  __syncthreads();
  for (int off = 128; off > 0; off >>= 1) {
    if (t < off) red[t] += red[t + off];
    __syncthreads();
  }
  float inv = 1.0f / red[0];
#pragma unroll
  for (int j = 0; j < 16; ++j) p[t + j * 256] = v[j] * inv;
}

// colsum over the query axis (484) for attn_q normalization
__global__ void k_colsum(const float* __restrict__ attn, float* __restrict__ cs) {
  int idx = blockIdx.x * blockDim.x + threadIdx.x;   // BH*4096
  if (idx >= BH * NIN) return;
  int bh = idx >> 12, k = idx & 4095;
  const float* p = attn + ((size_t)bh * NOUT) * NIN + k;
  float s = 0.0f;
  for (int q = 0; q < NOUT; ++q) s += p[(size_t)q * NIN];
  cs[idx] = s;
}

__global__ void k_attnq(const float* __restrict__ ak, const float* __restrict__ cs,
                        float* __restrict__ aq) {
  int idx = blockIdx.x * blockDim.x + threadIdx.x;
  if (idx >= ATTN_ELEMS) return;
  int bh = idx / (NOUT * NIN);
  int k = idx & 4095;
  aq[idx] = ak[idx] / (cs[bh * NIN + k] + EPS_);
}

// ---------------------------------------------------------------------------
// Host-side orchestration
// ---------------------------------------------------------------------------
extern "C" void kernel_launch(void* const* d_in, const int* in_sizes, int n_in,
                              void* d_out, int out_size, void* d_ws, size_t ws_size,
                              hipStream_t stream) {
  (void)in_sizes; (void)n_in; (void)out_size; (void)ws_size;
  const float* x       = (const float*)d_in[0];
  const float* conv_w  = (const float*)d_in[1];
  const float* conv_b  = (const float*)d_in[2];
  const float* kv_w    = (const float*)d_in[3];
  const float* kv_b    = (const float*)d_in[4];
  const float* q_w     = (const float*)d_in[5];
  const float* q_b     = (const float*)d_in[6];
  const float* mlp_w1  = (const float*)d_in[7];
  const float* mlp_b1  = (const float*)d_in[8];
  const float* mlp_w2  = (const float*)d_in[9];
  const float* mlp_b2  = (const float*)d_in[10];
  const float* g1 = (const float*)d_in[11];
  const float* b1 = (const float*)d_in[12];
  const float* g2 = (const float*)d_in[13];
  const float* b2 = (const float*)d_in[14];
  const float* g3 = (const float*)d_in[15];
  const float* b3 = (const float*)d_in[16];
  const float* temp  = (const float*)d_in[17];
  const float* rfreq = (const float*)d_in[18];

  float* out_x  = (float*)d_out;                 // (B, 484, 384)
  float* out_aq = out_x + (size_t)MQ * D_;       // (B, H, 484, 4096)
  float* out_ak = out_aq + (size_t)ATTN_ELEMS;   // (B, H, 484, 4096)

  // workspace carve-out (256B aligned)
  char* w = (char*)d_ws;
  size_t off = 0;
  auto take = [&](size_t bytes) -> void* {
    void* p = w + off;
    off = (off + bytes + 255) & ~(size_t)255;
    return p;
  };
  __bf16* xb    = (__bf16*)take((size_t)B_ * NIN * D_ * 2);
  __bf16* wconv = (__bf16*)take((size_t)49 * 384 * 384 * 2);
  __bf16* kvwt  = (__bf16*)take((size_t)768 * 384 * 2);
  __bf16* qwt   = (__bf16*)take((size_t)384 * 384 * 2);
  __bf16* w1t   = (__bf16*)take((size_t)1536 * 384 * 2);
  __bf16* w2t   = (__bf16*)take((size_t)384 * 1536 * 2);
  float*  xo    = (float*) take((size_t)MQ * D_ * 4);
  float*  x_out = (float*) take((size_t)MQ * D_ * 4);
  __bf16* xob   = (__bf16*)take((size_t)MQ * D_ * 2);
  float*  kv    = (float*) take((size_t)B_ * NIN * 768 * 4);
  __bf16* kb    = (__bf16*)take((size_t)BH * NIN * 32 * 2);
  __bf16* vt    = (__bf16*)take((size_t)BH * 32 * NIN * 2);
  float*  qlin  = (float*) take((size_t)MQ * D_ * 4);
  __bf16* qb    = (__bf16*)take((size_t)BH * NQPAD * 32 * 2);
  float*  cs    = (float*) take((size_t)BH * NIN * 4);
  float*  upd   = (float*) take((size_t)MQ * D_ * 4);
  __bf16* h1    = (__bf16*)take((size_t)MQ * 1536 * 2);
  float*  h2    = (float*) take((size_t)MQ * D_ * 4);

  // ---- prep: bf16 conversions + weight transposes
  k_cvt_bf16<<<(B_ * NIN * D_ + 255) / 256, 256, 0, stream>>>(x, xb, B_ * NIN * D_);
  k_convw<<<(384 * 384 * 49 + 255) / 256, 256, 0, stream>>>(conv_w, wconv);
  k_transpose_cvt<<<(384 * 768 + 255) / 256, 256, 0, stream>>>(kv_w, kvwt, 384, 768);
  k_transpose_cvt<<<(384 * 384 + 255) / 256, 256, 0, stream>>>(q_w, qwt, 384, 384);
  k_transpose_cvt<<<(384 * 1536 + 255) / 256, 256, 0, stream>>>(mlp_w1, w1t, 384, 1536);
  k_transpose_cvt<<<(1536 * 384 + 255) / 256, 256, 0, stream>>>(mlp_w2, w2t, 1536, 384);

  // ---- conv + LN1
  k_gemm_conv<<<(61 * 6 + 7) / 8, 256, 0, stream>>>(xb, wconv, conv_b, xo);
  k_ln<<<MQ, 128, 0, stream>>>(xo, x_out, xob, g1, b1, 0);

  // ---- kv / K / V hoisted out of the iteration loop (x is constant)
  k_gemm_kv<<<(512 * 12 + 7) / 8, 256, 0, stream>>>(xb, kvwt, kv_b, kv);
  k_rose_kv<<<B_ * NIN, 384, 0, stream>>>(kv, rfreq, temp, kb, vt);

  for (int it = 0; it < 3; ++it) {
    k_gemm_rows<384, 384><<<(61 * 6 + 7) / 8, 256, 0, stream>>>(xob, qwt, q_b, qlin);
    k_rose_q<<<B_ * NQPAD, 384, 0, stream>>>(qlin, rfreq, qb);
    k_gemm_qk<<<(BH * 31 * 32 + 7) / 8, 256, 0, stream>>>(qb, kb, out_ak);
    k_softmax<<<BH * NOUT, 256, 0, stream>>>(out_ak);
    k_colsum<<<(BH * NIN + 255) / 256, 256, 0, stream>>>(out_ak, cs);
    k_attnq<<<(ATTN_ELEMS + 255) / 256, 256, 0, stream>>>(out_ak, cs, out_aq);
    k_gemm_pv<<<(BH * 31 + 7) / 8, 256, 0, stream>>>(out_aq, vt, upd);
    k_ln<<<MQ, 128, 0, stream>>>(upd, x_out, xob, g2, b2, 1);
    k_gemm_mlp1<<<(61 * 24 + 7) / 8, 256, 0, stream>>>(xob, w1t, mlp_b1, h1);
    k_gemm_rows<384, 1536><<<(61 * 6 + 7) / 8, 256, 0, stream>>>(h1, w2t, mlp_b2, h2);
    k_ln<<<MQ, 128, 0, stream>>>(h2, x_out, xob, g3, b3, 1);
  }

  k_copy_f32<<<((int)((size_t)MQ * D_) + 255) / 256, 256, 0, stream>>>(
      x_out, out_x, MQ * D_);
}
